// ViT_sparse_40029095198750
// MI455X (gfx1250) — matrix operand order repaired and checked by reading, not compiled, not run on hardware
//
#include <hip/hip_runtime.h>
#include <hip/hip_bf16.h>
#include <math.h>

typedef __bf16 v16bf __attribute__((ext_vector_type(16)));
typedef __bf16 v8bf  __attribute__((ext_vector_type(8)));
typedef float  v8f   __attribute__((ext_vector_type(8)));
typedef int    i32x4 __attribute__((ext_vector_type(4)));
typedef int    i32x8 __attribute__((ext_vector_type(8)));

#define NB 16
#define NC 3
#define HH 384
#define GG 24
#define NPATCH 576
#define SEQ 577
#define DM 768
#define FFD 3072
#define NHEAD 12
#define NLAYER 12
#define NCLS 1000
#define KSEL 288
#define SS 289
#define HDIM 64

__device__ __forceinline__ float gelu_f(float x) {
  return 0.5f * x * (1.0f + erff(x * 0.70710678118654752f));
}

__device__ __forceinline__ v8f wmma_bf16(v16bf a, v16bf b, v8f c) {
  return __builtin_amdgcn_wmma_f32_16x16x32_bf16(false, a, false, b, (short)0, c, false, false);
}

// ---------------------------------------------------------------------------
// Generic GEMM: C = act(A(MxK) * W(KxN) + bias) [+ residual]; bf16 WMMA, f32 acc
// Block: 256 threads (8 waves). Tile: 128x128, K-step 32. Wave tile: 32x64.
// A tile staged by the Tensor Data Mover (tensor_load_to_lds, TENSORcnt);
// B tile staged by the waves (f32 -> bf16, column-major for contiguous frags).
// ---------------------------------------------------------------------------
template <int ACT>
__global__ void __launch_bounds__(256)
gemm_wmma_kernel(const float* __restrict__ A, const float* __restrict__ W,
                 const float* __restrict__ bias, const float* __restrict__ res,
                 float* __restrict__ Cout, int M, int Kd, int N) {
  __shared__ float  sA[128 * 32];      // f32 row-major [row][k], written by TDM
  __shared__ __bf16 sB[128 * 32];      // bf16 col-major [n][k]
  const int tid  = threadIdx.x;
  const int lane = tid & 31;
  const int wid  = tid >> 5;
  const int half = lane >> 4;
  const int l16  = lane & 15;
  const int wm = wid & 3, wn = wid >> 2;          // 4x2 waves -> 128x128
  const int mBase = blockIdx.y * 128;
  const int nBase = blockIdx.x * 128;

  // Constant part of the TDM descriptor (uniform, lives in SGPRs)
  const unsigned ldsoff = (unsigned)(unsigned long long)(uintptr_t)&sA[0];
  const unsigned long long abase =
      (unsigned long long)(uintptr_t)A + ((size_t)mBase * Kd) * 4ull;
  const unsigned td0 = (unsigned)Kd;              // tensor_dim0 (elements)
  const unsigned td1 = (unsigned)(M - mBase);     // remaining rows (OOB -> 0)
  const unsigned long long st0 = (unsigned long long)Kd;

  v8f acc[2][4];
  for (int i = 0; i < 2; i++)
    for (int j = 0; j < 4; j++)
      for (int e = 0; e < 8; e++) acc[i][j][e] = 0.f;

  for (int kt = 0; kt < Kd; kt += 32) {
    // ---- issue TDM load of the A tile (one wave, DMA writes LDS directly)
    if (wid == 0) {
      unsigned long long gaddr = abase + (unsigned long long)kt * 4ull;
      i32x4 g0;
      g0[0] = 1;                                   // count=1, user descriptor
      g0[1] = (int)ldsoff;                         // lds_addr (bytes)
      g0[2] = (int)(unsigned)(gaddr & 0xffffffffull);
      g0[3] = (int)(((unsigned)(gaddr >> 32) & 0x1ffffffu) | (2u << 30)); // type=2
      i32x8 g1;
      g1[0] = (int)(2u << 16);                     // data_size=4B, no mask/flags
      g1[1] = (int)((td0 & 0xffffu) << 16);        // tensor_dim0 lo
      g1[2] = (int)(((td1 & 0xffffu) << 16) | (td0 >> 16));
      g1[3] = (int)((32u << 16) | (td1 >> 16));    // tile_dim0=32 | dim1 hi
      g1[4] = (int)128u;                           // tile_dim1=128, tile_dim2=0
      g1[5] = (int)(unsigned)(st0 & 0xffffffffull);// tensor_dim0_stride lo
      g1[6] = (int)(unsigned)((st0 >> 32) & 0xffffull);
      g1[7] = 0;
      asm volatile("tensor_load_to_lds %0, %1" :: "s"(g0), "s"(g1) : "memory");
    }

    // ---- B tile: f32 -> bf16, column-major (128 cols x 32 k)
    for (int i = 0; i < 16; i++) {                 // 128*32 / 256
      int idx = tid + i * 256;
      int k = idx >> 7, n = idx & 127;
      int gn = nBase + n;
      float v = (gn < N) ? W[(size_t)(kt + k) * N + gn] : 0.f;
      sB[n * 32 + k] = (__bf16)v;
    }
    // prefetch next weight K-tile into L2 while this tile computes
    if (kt + 32 < Kd) {
      int pn = nBase + ((tid & 7) << 4);
      if (pn < N)
        __builtin_prefetch(&W[(size_t)(kt + 32 + (tid >> 3)) * N + pn], 0, 1);
    }

    if (wid == 0) __builtin_amdgcn_s_wait_tensorcnt(0);
    __syncthreads();

    // ---- fragments
    v16bf af[2], bfr[4];
    for (int ti = 0; ti < 2; ti++) {
      const float* bp = &sA[(wm * 32 + ti * 16 + l16) * 32];
      float4 f0 = *(const float4*)(bp + half * 8);
      float4 f1 = *(const float4*)(bp + half * 8 + 4);
      float4 f2 = *(const float4*)(bp + 16 + half * 8);
      float4 f3 = *(const float4*)(bp + 16 + half * 8 + 4);
      v16bf a;
      a[0]  = (__bf16)f0.x; a[1]  = (__bf16)f0.y; a[2]  = (__bf16)f0.z; a[3]  = (__bf16)f0.w;
      a[4]  = (__bf16)f1.x; a[5]  = (__bf16)f1.y; a[6]  = (__bf16)f1.z; a[7]  = (__bf16)f1.w;
      a[8]  = (__bf16)f2.x; a[9]  = (__bf16)f2.y; a[10] = (__bf16)f2.z; a[11] = (__bf16)f2.w;
      a[12] = (__bf16)f3.x; a[13] = (__bf16)f3.y; a[14] = (__bf16)f3.z; a[15] = (__bf16)f3.w;
      af[ti] = a;
    }
    for (int tj = 0; tj < 4; tj++) {
      const __bf16* base = &sB[(wn * 64 + tj * 16 + l16) * 32];
      v8bf lo = *(const v8bf*)(base + half * 8);
      v8bf hi = *(const v8bf*)(base + 16 + half * 8);
      bfr[tj] = __builtin_shufflevector(lo, hi, 0, 1, 2, 3, 4, 5, 6, 7,
                                        8, 9, 10, 11, 12, 13, 14, 15);
    }
    for (int ti = 0; ti < 2; ti++)
      for (int tj = 0; tj < 4; tj++)
        acc[ti][tj] = wmma_bf16(af[ti], bfr[tj], acc[ti][tj]);
    __syncthreads();
  }

  for (int ti = 0; ti < 2; ti++)
    for (int tj = 0; tj < 4; tj++) {
      int gcol = nBase + wn * 64 + tj * 16 + l16;
      if (gcol >= N) continue;
      float bv = bias ? bias[gcol] : 0.f;
      for (int r = 0; r < 8; r++) {
        int grow = mBase + wm * 32 + ti * 16 + (half ? r + 8 : r);
        if (grow >= M) continue;
        float v = acc[ti][tj][r] + bv;
        if (ACT == 1) v = gelu_f(v);
        size_t o = (size_t)grow * N + gcol;
        if (res) v += res[o];
        Cout[o] = v;
      }
    }
}

// ---------------------------------------------------------------------------
// Flash-style attention. One wave per (batch, head, 16-query tile).
// qkv: (B*S, 3D) f32, q|k|v split at D. out o: (B*S, D).
// ---------------------------------------------------------------------------
__global__ void __launch_bounds__(32)
attn_kernel(const float* __restrict__ qkv, float* __restrict__ o) {
  const int qt   = blockIdx.x;   // 0..18
  const int h    = blockIdx.y;
  const int b    = blockIdx.z;
  const int lane = threadIdx.x & 31;
  const int half = lane >> 4;
  const int l16  = lane & 15;
  const size_t bS = (size_t)b * SS;
  const int h64 = h * HDIM;
  __shared__ __bf16 lds_p[16 * 32];

  v16bf qa[2];
  {
    int q = qt * 16 + l16;
    for (int kk = 0; kk < 2; kk++) {
      if (q < SS) {
        const float* qp = qkv + (bS + q) * (3 * DM) + h64 + kk * 32;
        for (int e = 0; e < 8; e++) {
          qa[kk][e]     = (__bf16)qp[half * 16 + e];
          qa[kk][8 + e] = (__bf16)qp[half * 16 + 8 + e];
        }
      } else {
        for (int e = 0; e < 16; e++) qa[kk][e] = (__bf16)0.f;
      }
    }
  }

  v8f oacc[4];
  for (int t = 0; t < 4; t++)
    for (int e = 0; e < 8; e++) oacc[t][e] = 0.f;
  float m_i[8], l_i[8];
  for (int r = 0; r < 8; r++) { m_i[r] = -3.0e38f; l_i[r] = 0.f; }

  for (int kb = 0; kb < 320; kb += 32) {           // 10 tiles of 32 keys
    v8f s[2];
    for (int st = 0; st < 2; st++) {
      int key = kb + st * 16 + l16;
      v8f z;
      for (int e = 0; e < 8; e++) z[e] = 0.f;
      for (int kk = 0; kk < 2; kk++) {
        v16bf kf;
        if (key < SS) {
          const float* kp = qkv + (bS + key) * (3 * DM) + DM + h64 + kk * 32 + half * 16;
          for (int e = 0; e < 16; e++) kf[e] = (__bf16)kp[e];
        } else {
          for (int e = 0; e < 16; e++) kf[e] = (__bf16)0.f;
        }
        z = wmma_bf16(qa[kk], kf, z);
      }
      for (int e = 0; e < 8; e++) z[e] *= 0.125f;  // 1/sqrt(64)
      if (key >= SS)
        for (int e = 0; e < 8; e++) z[e] = -1.0e30f;
      s[st] = z;
    }

    for (int r = 0; r < 8; r++) {
      float mx = fmaxf(s[0][r], s[1][r]);
      for (int off = 8; off >= 1; off >>= 1) mx = fmaxf(mx, __shfl_xor(mx, off, 16));
      float mnew  = fmaxf(m_i[r], mx);
      float alpha = __expf(m_i[r] - mnew);
      float p0 = __expf(s[0][r] - mnew);
      float p1 = __expf(s[1][r] - mnew);
      s[0][r] = p0; s[1][r] = p1;
      float rs = p0 + p1;
      for (int off = 8; off >= 1; off >>= 1) rs += __shfl_xor(rs, off, 16);
      l_i[r] = l_i[r] * alpha + rs;
      m_i[r] = mnew;
      for (int t = 0; t < 4; t++) oacc[t][r] *= alpha;
    }

    __syncthreads();
    for (int st = 0; st < 2; st++)
      for (int r = 0; r < 8; r++) {
        int row = half ? r + 8 : r;
        lds_p[row * 32 + st * 16 + l16] = (__bf16)s[st][r];
      }
    __syncthreads();
    v16bf pf;
    {
      const __bf16* base = &lds_p[l16 * 32];
      for (int e = 0; e < 8; e++) {
        pf[e]     = base[half * 16 + e];
        pf[8 + e] = base[half * 16 + 8 + e];
      }
    }
    for (int t = 0; t < 4; t++) {
      int d = t * 16 + l16;
      v16bf vf;
      for (int e = 0; e < 16; e++) {
        int key = kb + half * 16 + e;
        vf[e] = (key < SS)
                    ? (__bf16)qkv[(bS + key) * (3 * DM) + 2 * DM + h64 + d]
                    : (__bf16)0.f;
      }
      oacc[t] = wmma_bf16(pf, vf, oacc[t]);
    }
    __syncthreads();
  }

  for (int t = 0; t < 4; t++) {
    int d = t * 16 + l16;
    for (int r = 0; r < 8; r++) {
      int q = qt * 16 + (half ? r + 8 : r);
      if (q < SS) o[(bS + q) * DM + h64 + d] = oacc[t][r] / l_i[r];
    }
  }
}

// ---------------------------------------------------------------------------
__global__ void __launch_bounds__(256)
ln_kernel(const float* __restrict__ in, float* __restrict__ out,
          const float* __restrict__ g, const float* __restrict__ bta, long rstride) {
  const int row = blockIdx.x, tid = threadIdx.x;
  const float* p = in + (size_t)row * rstride;
  float a0 = p[tid], a1 = p[tid + 256], a2 = p[tid + 512];
  __shared__ float s1[256], s2[256];
  s1[tid] = a0 + a1 + a2;
  s2[tid] = a0 * a0 + a1 * a1 + a2 * a2;
  __syncthreads();
  for (int s = 128; s > 0; s >>= 1) {
    if (tid < s) { s1[tid] += s1[tid + s]; s2[tid] += s2[tid + s]; }
    __syncthreads();
  }
  float mean = s1[0] * (1.f / 768.f);
  float var  = s2[0] * (1.f / 768.f) - mean * mean;
  float rstd = rsqrtf(var + 1e-6f);
  float* q = out + (size_t)row * 768;
  q[tid]       = (a0 - mean) * rstd * g[tid]       + bta[tid];
  q[tid + 256] = (a1 - mean) * rstd * g[tid + 256] + bta[tid + 256];
  q[tid + 512] = (a2 - mean) * rstd * g[tid + 512] + bta[tid + 512];
}

__global__ void __launch_bounds__(256)
im2col_kernel(const float* __restrict__ x, float* __restrict__ out) {
  int pr = blockIdx.x;                      // b*576 + p
  int b = pr / NPATCH, p = pr % NPATCH;
  int gy = p / GG, gx = p % GG;
  for (int i = 0; i < 3; i++) {
    int e = threadIdx.x + i * 256;
    int c = e >> 8, rem = e & 255;
    int py = rem >> 4, px = rem & 15;
    out[(size_t)pr * 768 + e] =
        x[(((size_t)b * NC + c) * HH + gy * 16 + py) * HH + gx * 16 + px];
  }
}

__global__ void __launch_bounds__(256)
transpose_pw_kernel(const float* __restrict__ w, float* __restrict__ wt) {
  int idx = blockIdx.x * 256 + threadIdx.x;
  int k = idx / 768, n = idx % 768;
  wt[(size_t)k * 768 + n] = w[(size_t)n * 768 + k];
}

__global__ void __launch_bounds__(576)
pool_topk_kernel(const float* __restrict__ x, int* __restrict__ idx_out) {
  const int b = blockIdx.x, tid = threadIdx.x;
  __shared__ float key[1024];
  __shared__ int   val[1024];
  if (tid < NPATCH) {
    int gy = tid / GG, gx = tid % GG;
    const float* base = x + (size_t)b * NC * HH * HH;   // channel 0
    float s = 0.f;
    for (int py = 0; py < 16; py++) {
      const float* rowp = base + (size_t)(gy * 16 + py) * HH + gx * 16;
      for (int px = 0; px < 16; px++) s += rowp[px];
    }
    key[tid] = -fabsf(s * (1.f / 256.f));
    val[tid] = tid;
  }
  for (int i = tid + 576; i < 1024; i += 576) { key[i] = 3.0e38f; val[i] = 1 << 30; }
  __syncthreads();
  for (int k = 2; k <= 1024; k <<= 1) {
    for (int j = k >> 1; j > 0; j >>= 1) {
      for (int i = tid; i < 1024; i += 576) {
        int l = i ^ j;
        if (l > i) {
          bool up = ((i & k) == 0);
          float ki = key[i], kl = key[l];
          if ((ki > kl) == up) {
            key[i] = kl; key[l] = ki;
            int t = val[i]; val[i] = val[l]; val[l] = t;
          }
        }
      }
      __syncthreads();
    }
  }
  if (tid < KSEL) idx_out[b * KSEL + tid] = val[tid];
}

__global__ void __launch_bounds__(256)
gather_kernel(const float* __restrict__ patch_out, const int* __restrict__ idx,
              const float* __restrict__ cls_tok, const float* __restrict__ pos_emb,
              float* __restrict__ h) {
  int row = blockIdx.x;                 // b*S + s
  int b = row / SS, s = row % SS;
  float* hp = h + (size_t)row * DM;
  if (s == 0) {
    for (int i = 0; i < 3; i++) {
      int d = threadIdx.x + i * 256;
      hp[d] = cls_tok[d] + pos_emb[d];
    }
  } else {
    int p = idx[b * KSEL + (s - 1)];
    const float* src = patch_out + ((size_t)b * NPATCH + p) * DM;
    const float* pe  = pos_emb + (size_t)(1 + p) * DM;
    for (int i = 0; i < 3; i++) {
      int d = threadIdx.x + i * 256;
      hp[d] = src[d] + pe[d];
    }
  }
}

extern "C" void kernel_launch(void* const* d_in, const int* in_sizes, int n_in,
                              void* d_out, int out_size, void* d_ws, size_t ws_size,
                              hipStream_t stream) {
  const float* x       = (const float*)d_in[0];
  const float* patch_w = (const float*)d_in[1];
  const float* patch_b = (const float*)d_in[2];
  const float* cls_tok = (const float*)d_in[3];
  const float* pos_emb = (const float*)d_in[4];
  const float* ln1_g   = (const float*)d_in[5];
  const float* ln1_b   = (const float*)d_in[6];
  const float* Wqkv    = (const float*)d_in[7];
  const float* bqkv    = (const float*)d_in[8];
  const float* Wo      = (const float*)d_in[9];
  const float* bo      = (const float*)d_in[10];
  const float* ln2_g   = (const float*)d_in[11];
  const float* ln2_b   = (const float*)d_in[12];
  const float* W1      = (const float*)d_in[13];
  const float* b1      = (const float*)d_in[14];
  const float* W2      = (const float*)d_in[15];
  const float* b2      = (const float*)d_in[16];
  const float* lnf_g   = (const float*)d_in[17];
  const float* lnf_b   = (const float*)d_in[18];
  const float* fc_w    = (const float*)d_in[19];
  const float* fc_b    = (const float*)d_in[20];
  float* out = (float*)d_out;
  (void)in_sizes; (void)n_in; (void)out_size; (void)ws_size;

  char* ws = (char*)d_ws;
  size_t off = 0;
  auto alloc = [&](size_t nelem) -> float* {
    float* p = (float*)(ws + off);
    off += ((nelem * 4) + 255) & ~(size_t)255;
    return p;
  };
  const int M = NB * SS;                                  // 4624
  float* pmat = alloc((size_t)NB * NPATCH * DM);
  float* pout = alloc((size_t)NB * NPATCH * DM);
  float* pwT  = alloc((size_t)DM * DM);
  float* h    = alloc((size_t)M * DM);
  float* y    = alloc((size_t)M * DM);
  float* qkv  = alloc((size_t)M * 3 * DM);
  float* attn = alloc((size_t)M * DM);
  float* ff   = alloc((size_t)M * FFD);
  float* clsy = alloc((size_t)NB * DM);
  int* idx = (int*)(ws + off);
  off += ((size_t)NB * KSEL * 4 + 255) & ~(size_t)255;

  auto gemm = [&](const float* A, const float* W, const float* bias,
                  const float* res, float* C, int m, int k, int n, int act) {
    dim3 g((n + 127) / 128, (m + 127) / 128);
    if (act)
      gemm_wmma_kernel<1><<<g, 256, 0, stream>>>(A, W, bias, res, C, m, k, n);
    else
      gemm_wmma_kernel<0><<<g, 256, 0, stream>>>(A, W, bias, res, C, m, k, n);
  };

  im2col_kernel<<<NB * NPATCH, 256, 0, stream>>>(x, pmat);
  transpose_pw_kernel<<<(DM * DM) / 256, 256, 0, stream>>>(patch_w, pwT);
  gemm(pmat, pwT, patch_b, nullptr, pout, NB * NPATCH, DM, DM, 0);
  pool_topk_kernel<<<NB, 576, 0, stream>>>(x, idx);
  gather_kernel<<<NB * SS, 256, 0, stream>>>(pout, idx, cls_tok, pos_emb, h);

  for (int L = 0; L < NLAYER; L++) {
    ln_kernel<<<M, 256, 0, stream>>>(h, y, ln1_g + L * DM, ln1_b + L * DM, DM);
    gemm(y, Wqkv + (size_t)L * DM * 3 * DM, bqkv + (size_t)L * 3 * DM,
         nullptr, qkv, M, DM, 3 * DM, 0);
    attn_kernel<<<dim3(19, NHEAD, NB), 32, 0, stream>>>(qkv, attn);
    gemm(attn, Wo + (size_t)L * DM * DM, bo + (size_t)L * DM, h, h, M, DM, DM, 0);
    ln_kernel<<<M, 256, 0, stream>>>(h, y, ln2_g + L * DM, ln2_b + L * DM, DM);
    gemm(y, W1 + (size_t)L * DM * FFD, b1 + (size_t)L * FFD, nullptr, ff, M, DM, FFD, 1);
    gemm(ff, W2 + (size_t)L * FFD * DM, b2 + (size_t)L * DM, h, h, M, FFD, DM, 0);
  }

  ln_kernel<<<NB, 256, 0, stream>>>(h, clsy, lnf_g, lnf_b, (long)SS * DM);
  gemm(clsy, fc_w, fc_b, nullptr, out, NB, DM, NCLS, 0);
}
